// OuterProductMean_63608465654535
// MI455X (gfx1250) — matrix operand order repaired and checked by reading, not compiled, not run on hardware
//
#include <hip/hip_runtime.h>
#include <hip/hip_bf16.h>

// ---------------------------------------------------------------------------
// OuterProductMean fused for MI455X (gfx1250, wave32, WMMA bf16 16x16x32)
//   msa (1,64,384,256) -LN-> x -proj-> a,b  (stored transposed: [l][c][s] bf16)
//   pair[i,j,p] = sum_cd [ (1/64) sum_s a[s,i,c] b[s,j,d] ] * Wout[c*32+d,p] + bout[p]
// outer tile never touches HBM: staged in LDS per (i, 16-j) block.
// All WMMA fragments are per-lane contiguous -> b128 loads (no scalar gathers).
// ---------------------------------------------------------------------------

typedef __attribute__((ext_vector_type(16))) __bf16 v16bf;
typedef __attribute__((ext_vector_type(8)))  __bf16 bf16x8;
typedef __attribute__((ext_vector_type(8)))  float  v8f;

#define D_MSA   256
#define N_SEQ   64
#define L_RES   384
#define N_OUT   32
#define D_PAIR  128
#define K_CD    (N_OUT * N_OUT)   // 1024

// ---------------- WMMA fragment loads (ISA 7.12.2 layouts) ----------------
// A matrix 16x32 (MxK) bf16 from row-major source, leading dim `ld`:
//   lane (m = lane&15, half = lane>>4) holds two contiguous 8-elem runs:
//   f[0..7]  = A[m][ 8*half      .. +7]   (VGPR 0..3)
//   f[8..15] = A[m][16 + 8*half .. +7]    (VGPR 4..7)
template <typename PTR>
__device__ __forceinline__ v16bf load_a_frag(PTR base, int ld) {
    const int lane = threadIdx.x & 31;
    const int m    = lane & 15;
    const int half = lane >> 4;
    bf16x8 lo = *(const bf16x8*)(&base[m * ld + 8 * half]);
    bf16x8 hi = *(const bf16x8*)(&base[m * ld + 16 + 8 * half]);
    v16bf f;
#pragma unroll
    for (int e = 0; e < 8; ++e) { f[e] = lo[e]; f[8 + e] = hi[e]; }
    return f;
}

// B matrix 32x16 (KxN) bf16 from an N-MAJOR source (Bt row-major N x K, ld):
//   lane (n = lane&15, half = lane>>4): f[0..7] = Bt[n][8*half .. +7],
//   f[8..15] = Bt[n][16 + 8*half .. +7]  (same K order as load_a_frag).
template <typename PTR>
__device__ __forceinline__ v16bf load_bt_frag(PTR base, int ld) {
    const int lane = threadIdx.x & 31;
    const int n    = lane & 15;
    const int half = lane >> 4;
    bf16x8 lo = *(const bf16x8*)(&base[n * ld + 8 * half]);
    bf16x8 hi = *(const bf16x8*)(&base[n * ld + 16 + 8 * half]);
    v16bf f;
#pragma unroll
    for (int e = 0; e < 8; ++e) { f[e] = lo[e]; f[8 + e] = hi[e]; }
    return f;
}

__device__ __forceinline__ v8f wmma_bf16(v16bf a, v16bf b, v8f c) {
    return __builtin_amdgcn_wmma_f32_16x16x32_bf16(
        /*neg_a=*/false, a, /*neg_b=*/false, b,
        /*c_mod=*/(short)0, c, /*reuse_a=*/false, /*reuse_b=*/false);
}

// ---------------------------------------------------------------------------
// Kernel 1: LayerNorm + dual projection.  One block (256 thr) per (s,l) row.
// Writes a_ws/b_ws TRANSPOSED: [l][c][s] bf16 (per l: 32 x 64 row-major).
// ---------------------------------------------------------------------------
__global__ void ln_proj_kernel(const float* __restrict__ msa,
                               const float* __restrict__ gamma,
                               const float* __restrict__ beta,
                               const float* __restrict__ Wa,
                               const float* __restrict__ ba,
                               const float* __restrict__ Wb,
                               const float* __restrict__ bb,
                               __bf16* __restrict__ a_ws,
                               __bf16* __restrict__ b_ws) {
    __shared__ float xs[D_MSA];
    __shared__ float red[8];
    __shared__ float parts[8][N_OUT];

    const int row = blockIdx.x;            // row = s*384 + l
    const int s   = row / L_RES;
    const int l   = row % L_RES;
    const int t   = threadIdx.x;           // 0..255 == d
    const int wid = t >> 5;

    const float x = msa[(size_t)row * D_MSA + t];

    // mean
    float v = x;
#pragma unroll
    for (int off = 16; off > 0; off >>= 1) v += __shfl_xor(v, off, 32);
    if ((t & 31) == 0) red[wid] = v;
    __syncthreads();
    float tot = 0.f;
#pragma unroll
    for (int k = 0; k < 8; ++k) tot += red[k];
    const float mu = tot * (1.0f / D_MSA);
    __syncthreads();

    // variance
    const float dx = x - mu;
    v = dx * dx;
#pragma unroll
    for (int off = 16; off > 0; off >>= 1) v += __shfl_xor(v, off, 32);
    if ((t & 31) == 0) red[wid] = v;
    __syncthreads();
    tot = 0.f;
#pragma unroll
    for (int k = 0; k < 8; ++k) tot += red[k];
    const float var = tot * (1.0f / D_MSA);

    xs[t] = dx * __frsqrt_rn(var + 1e-5f) * gamma[t] + beta[t];
    __syncthreads();

    // projections: thread t -> (c = t&31, d-chunk q = t>>5 of 32 elems)
    const int c = t & 31;
    const int q = t >> 5;
    float pa = 0.f, pb = 0.f;
#pragma unroll 8
    for (int d = q * 32; d < q * 32 + 32; ++d) {
        const float xv = xs[d];
        pa += xv * Wa[d * N_OUT + c];
        pb += xv * Wb[d * N_OUT + c];
    }
    parts[q][c] = pa;
    __syncthreads();
    if (t < N_OUT) {
        float acc = ba[t];
#pragma unroll
        for (int k = 0; k < 8; ++k) acc += parts[k][t];
        a_ws[((size_t)l * N_OUT + t) * N_SEQ + s] = (__bf16)acc;   // [l][c][s]
    }
    __syncthreads();
    parts[q][c] = pb;
    __syncthreads();
    if (t < N_OUT) {
        float acc = bb[t];
#pragma unroll
        for (int k = 0; k < 8; ++k) acc += parts[k][t];
        b_ws[((size_t)l * N_OUT + t) * N_SEQ + s] = (__bf16)acc;   // [l][d][s]
    }
}

// ---------------------------------------------------------------------------
// Kernel 2: Wout fp32 [cd][p] -> bf16 TRANSPOSED [p][cd] (L2-resident, 256 KB)
// ---------------------------------------------------------------------------
__global__ void wout_cvt_kernel(const float* __restrict__ w, __bf16* __restrict__ o) {
    const int idx = blockIdx.x * blockDim.x + threadIdx.x;   // over cd*128 + p
    const int k = idx >> 7;          // 0..1023
    const int p = idx & 127;         // 0..127
    o[(size_t)p * K_CD + k] = (__bf16)w[idx];
}

// ---------------------------------------------------------------------------
// Kernel 3: fused outer-product-mean + output projection.
//   grid = (24 j-tiles, 384 i), block = 256 (8 waves).
//   Phase 1: O_j(32x32) = A_i^T(32x64) * B_j(64x32) via WMMA, /64, -> LDS bf16
//   Phase 2: pair(16j x 128p) = outer_lds(16 x 1024) * Wout^T + bout
// ---------------------------------------------------------------------------
__global__ void opm_fused_kernel(const __bf16* __restrict__ a_ws,
                                 const __bf16* __restrict__ b_ws,
                                 const __bf16* __restrict__ wout_t,
                                 const float* __restrict__ bout,
                                 float* __restrict__ pair) {
    __shared__ alignas(16) __bf16 At[N_OUT * N_SEQ];      //  A_i^T : 32 x 64  (4 KB)
    __shared__ alignas(16) __bf16 outer_lds[16 * K_CD];   //  16 j  x 1024 cd (32 KB)

    const int jt   = blockIdx.x;                  // 0..23
    const int i    = blockIdx.y;                  // 0..383
    const int t    = threadIdx.x;
    const int w    = t >> 5;                      // wave 0..7
    const int lane = t & 31;
    const int half = lane >> 4;
    const int nloc = lane & 15;

    // stage A_i^T into LDS: straight 4 KB vector copy (a_ws is already [c][s])
    {
        const bf16x8* src = (const bf16x8*)(a_ws + (size_t)i * N_OUT * N_SEQ);
        bf16x8*       dst = (bf16x8*)At;
        dst[t] = src[t];                          // 256 thr x 16 B = 4 KB
    }
    __syncthreads();

    // ---------------- Phase 1: per-wave, 2 values of j ----------------
#pragma unroll
    for (int jj = 0; jj < 2; ++jj) {
        const int jl = 2 * w + jj;                       // 0..15
        const int j  = jt * 16 + jl;
        const __bf16* Bj = b_ws + (size_t)j * N_OUT * N_SEQ;   // [d][s], ld=64

        v8f acc00 = {}, acc01 = {}, acc10 = {}, acc11 = {};
#pragma unroll
        for (int ks = 0; ks < 2; ++ks) {                 // K = s, 2 x 32
            v16bf afr0 = load_a_frag(At + 0 * 16 * N_SEQ + ks * 32, N_SEQ);
            v16bf afr1 = load_a_frag(At + 1 * 16 * N_SEQ + ks * 32, N_SEQ);
            v16bf bfr0 = load_bt_frag(Bj + 0  * N_SEQ + ks * 32, N_SEQ);   // d 0..15
            v16bf bfr1 = load_bt_frag(Bj + 16 * N_SEQ + ks * 32, N_SEQ);   // d 16..31
            acc00 = wmma_bf16(afr0, bfr0, acc00);
            acc01 = wmma_bf16(afr0, bfr1, acc01);
            acc10 = wmma_bf16(afr1, bfr0, acc10);
            acc11 = wmma_bf16(afr1, bfr1, acc11);
        }
        // scatter D frags -> outer_lds[jl][c*32+d], scaled by 1/n_seqs
        const float scl = 1.0f / (float)N_SEQ;
#pragma unroll
        for (int vv = 0; vv < 8; ++vv) {
            const int m = vv + 8 * half;                 // row within 16
            const int n = nloc;                          // col within 16
            outer_lds[jl * K_CD + (0  + m) * N_OUT + (0  + n)] = (__bf16)(acc00[vv] * scl);
            outer_lds[jl * K_CD + (0  + m) * N_OUT + (16 + n)] = (__bf16)(acc01[vv] * scl);
            outer_lds[jl * K_CD + (16 + m) * N_OUT + (0  + n)] = (__bf16)(acc10[vv] * scl);
            outer_lds[jl * K_CD + (16 + m) * N_OUT + (16 + n)] = (__bf16)(acc11[vv] * scl);
        }
    }
    __syncthreads();

    // ---------------- Phase 2: wave w owns p-slice [16w, 16w+16) ----------------
    const int p0 = w * 16;
    v8f acc = {};
#pragma unroll 4
    for (int kt = 0; kt < K_CD / 32; ++kt) {             // 32 WMMA steps
        v16bf afr = load_a_frag(outer_lds + kt * 32, K_CD);                 // 16j x 32cd
        v16bf bfr = load_bt_frag(wout_t + (size_t)p0 * K_CD + kt * 32, K_CD);
        acc = wmma_bf16(afr, bfr, acc);
    }

    const int p  = p0 + nloc;
    const float bo = bout[p];
#pragma unroll
    for (int vv = 0; vv < 8; ++vv) {
        const int jl = vv + 8 * half;
        const int j  = jt * 16 + jl;
        pair[((size_t)i * L_RES + j) * D_PAIR + p] = acc[vv] + bo;
    }
}

// ---------------------------------------------------------------------------
extern "C" void kernel_launch(void* const* d_in, const int* in_sizes, int n_in,
                              void* d_out, int out_size, void* d_ws, size_t ws_size,
                              hipStream_t stream) {
    const float* msa   = (const float*)d_in[0];
    const float* gamma = (const float*)d_in[1];
    const float* beta  = (const float*)d_in[2];
    const float* Wa    = (const float*)d_in[3];
    const float* ba    = (const float*)d_in[4];
    const float* Wb    = (const float*)d_in[5];
    const float* bb    = (const float*)d_in[6];
    const float* Wout  = (const float*)d_in[7];
    const float* bout  = (const float*)d_in[8];
    float* pair        = (float*)d_out;

    __bf16* a_ws = (__bf16*)d_ws;                        // 384*32*64  [l][c][s]
    __bf16* b_ws = a_ws + (size_t)L_RES * N_OUT * N_SEQ; // 384*32*64  [l][d][s]
    __bf16* w_ws = b_ws + (size_t)L_RES * N_OUT * N_SEQ; // 128*1024   [p][cd]

    // K1: LayerNorm + projections (24576 rows)
    ln_proj_kernel<<<N_SEQ * L_RES, 256, 0, stream>>>(
        msa, gamma, beta, Wa, ba, Wb, bb, a_ws, b_ws);

    // K2: Wout -> bf16, transposed
    {
        const int n = K_CD * D_PAIR;
        wout_cvt_kernel<<<n / 256, 256, 0, stream>>>(Wout, w_ws);
    }

    // K3: fused outer product mean + projection
    dim3 grid(L_RES / 16, L_RES);   // (24 j-tiles, 384 i)
    opm_fused_kernel<<<grid, 256, 0, stream>>>(a_ws, b_ws, w_ws, bout, pair);
}
